// Double_SSM_Block_Custom_Channel_24713241821819
// MI455X (gfx1250) — matrix operand order repaired and checked by reading, not compiled, not run on hardware
//
#include <hip/hip_runtime.h>
#include <hip/hip_bf16.h>

// ---------------------------------------------------------------------------
// Double Mamba SSM block for MI455X (gfx1250, wave32).
// GEMMs use v_wmma_f32_16x16x32_bf16 (bf16 in, fp32 accumulate),
// 32x(16*TNT) register tile per wave for A/B reuse, global_prefetch ahead.
// ---------------------------------------------------------------------------

typedef __attribute__((ext_vector_type(16))) __bf16 v16bf;
typedef __attribute__((ext_vector_type(8)))  __bf16 v8bf;
typedef __attribute__((ext_vector_type(8)))  float  v8f;

#define MROWS   8192          // B*L = 8*1024
#define LSEQ    1024
#define BATCH   8
#define DMODEL  256
#define DIN     512
#define NSTATE  16
#define DTRANK  16

__device__ __forceinline__ __bf16 f2bf(float f) {
  unsigned u = __float_as_uint(f);
  unsigned r = (u + 0x7FFFu + ((u >> 16) & 1u)) >> 16;   // round-to-nearest-even
  unsigned short s = (unsigned short)r;
  return __builtin_bit_cast(__bf16, s);
}

__device__ __forceinline__ float silu_f(float x) {
  return x / (1.0f + __expf(-x));
}

// ---------------- f32 -> bf16 convert ----------------
__global__ void k_cvt_bf16(const float* __restrict__ src, __bf16* __restrict__ dst, int n) {
  int i = blockIdx.x * blockDim.x + threadIdx.x;
  if (i < n) dst[i] = f2bf(src[i]);
}

// ---------------- WMMA GEMM: C[M,N] = A[M,K] @ B[N,K]^T ----------------
// A row-major bf16, B row-major bf16 (weight layout, N x K), C row-major f32.
// One wave -> 32 x (16*TNT) output tile: 2 A-frags x TNT B-frags reused in
// registers, 2*TNT wmma per 32-wide K step.
template<int TNT>
__global__ void k_gemm_bf16_wmma(const __bf16* __restrict__ A,
                                 const __bf16* __restrict__ Bw,
                                 float* __restrict__ C,
                                 int M, int N, int K) {
  int wave  = blockIdx.x * (blockDim.x >> 5) + (threadIdx.x >> 5);
  int lane  = threadIdx.x & 31;
  int tilesN = N / (16 * TNT);
  int tilesM = M >> 5;                 // 32 rows per wave tile
  if (wave >= tilesM * tilesN) return;
  int tm = (wave / tilesN) << 5;
  int tn = (wave % tilesN) * (16 * TNT);

  int half = lane >> 4;        // lane group 0/1
  int r    = lane & 15;

  // A fragment layout (16-bit A 16x32): lane group 0 -> K runs {0..7,16..23},
  // lane group 1 -> {8..15,24..31}; two M-tiles (tm, tm+16).
  const __bf16* arow0 = A + (size_t)(tm + r) * K + half * 8;
  const __bf16* arow1 = arow0 + (size_t)16 * K;
  // B fragment (32x16, K-major): same K runs as A: group 0 -> {0..7,16..23}, group 1 -> {8..15,24..31}.
  const __bf16* brow[TNT];
#pragma unroll
  for (int t = 0; t < TNT; ++t)
    brow[t] = Bw + (size_t)(tn + 16 * t + r) * K + half * 8;

  v8f zero = {};
  v8f acc[2][TNT];
#pragma unroll
  for (int t = 0; t < TNT; ++t) { acc[0][t] = zero; acc[1][t] = zero; }

  for (int k0 = 0; k0 < K; k0 += 32) {
    // prefetch one K-step ahead (global_prefetch_b8, speculative)
    __builtin_prefetch(arow0 + k0 + 32, 0, 3);
    __builtin_prefetch(arow1 + k0 + 32, 0, 3);

    v8bf a0lo = *(const v8bf*)(arow0 + k0);
    v8bf a0hi = *(const v8bf*)(arow0 + k0 + 16);
    v8bf a1lo = *(const v8bf*)(arow1 + k0);
    v8bf a1hi = *(const v8bf*)(arow1 + k0 + 16);
    v16bf af0, af1;
#pragma unroll
    for (int i = 0; i < 8; ++i) {
      af0[i] = a0lo[i]; af0[i + 8] = a0hi[i];
      af1[i] = a1lo[i]; af1[i + 8] = a1hi[i];
    }
#pragma unroll
    for (int t = 0; t < TNT; ++t) {
      __builtin_prefetch(brow[t] + k0 + 32, 0, 3);
      v8bf blo = *(const v8bf*)(brow[t] + k0);
      v8bf bhi = *(const v8bf*)(brow[t] + k0 + 16);
      v16bf bf;
#pragma unroll
      for (int i = 0; i < 8; ++i) { bf[i] = blo[i]; bf[i + 8] = bhi[i]; }
      acc[0][t] = __builtin_amdgcn_wmma_f32_16x16x32_bf16(
          false, af0, false, bf, (short)0, acc[0][t], false, false);
      acc[1][t] = __builtin_amdgcn_wmma_f32_16x16x32_bf16(
          false, af1, false, bf, (short)0, acc[1][t], false, false);
    }
  }

  // C/D layout: VGPR i -> row + i + 8*half, col tn' + r
#pragma unroll
  for (int t = 0; t < TNT; ++t) {
    int col = tn + 16 * t + r;
#pragma unroll
    for (int i = 0; i < 8; ++i) {
      C[(size_t)(tm + half * 8 + i) * N + col]      = acc[0][t][i];
      C[(size_t)(tm + 16 + half * 8 + i) * N + col] = acc[1][t][i];
    }
  }
}

// ---------------- causal depthwise conv1d + SiLU ----------------
// xr: M x (2*DIN) f32 ; xz = columns [0,DIN). u (f32) and u_bf (bf16) out.
__global__ void k_conv_silu(const float* __restrict__ xr,
                            const float* __restrict__ wconv,   // DIN x 4
                            const float* __restrict__ bconv,   // DIN
                            float* __restrict__ u,
                            __bf16* __restrict__ u_bf) {
  int idx = blockIdx.x * blockDim.x + threadIdx.x;   // over MROWS*DIN
  if (idx >= MROWS * DIN) return;
  int d = idx & (DIN - 1);
  int m = idx >> 9;
  int l = m & (LSEQ - 1);
  float acc = bconv[d];
#pragma unroll
  for (int j = 0; j < 4; ++j) {
    int lj = l - 3 + j;
    if (lj >= 0)
      acc += xr[(size_t)(m - 3 + j) * (2 * DIN) + d] * wconv[d * 4 + j];
  }
  float s = silu_f(acc);
  u[idx]    = s;
  u_bf[idx] = f2bf(s);
}

// ---------------- delta = softplus(x_dbl[:, :16] @ w_dt^T + b_dt) ----------------
__global__ void k_delta(const float* __restrict__ xdbl,   // M x 48
                        const float* __restrict__ wdt,    // DIN x 16
                        const float* __restrict__ bdt,    // DIN
                        float* __restrict__ delta) {
  int idx = blockIdx.x * blockDim.x + threadIdx.x;   // over MROWS*DIN
  if (idx >= MROWS * DIN) return;
  int d = idx & (DIN - 1);
  int m = idx >> 9;
  const float* xd = xdbl + (size_t)m * 48;
  const float* w  = wdt + d * DTRANK;
  float acc = bdt[d];
#pragma unroll
  for (int r = 0; r < DTRANK; ++r) acc += xd[r] * w[r];
  float sp = (acc > 20.0f) ? acc : __logf(1.0f + __expf(acc));
  delta[idx] = sp;
}

// ---------------- selective scan: one thread per (b, d) channel ----------------
__global__ void k_scan(const float* __restrict__ delta,  // M x DIN
                       const float* __restrict__ u,      // M x DIN
                       const float* __restrict__ xdbl,   // M x 48 (B at 16, C at 32)
                       const float* __restrict__ A_log,  // DIN x 16
                       const float* __restrict__ Dp,     // DIN
                       float* __restrict__ y) {          // M x DIN
  int t = blockIdx.x * blockDim.x + threadIdx.x;         // 0 .. BATCH*DIN-1
  if (t >= BATCH * DIN) return;
  int d = t & (DIN - 1);
  int b = t >> 9;

  float An[NSTATE];
#pragma unroll
  for (int n = 0; n < NSTATE; ++n) An[n] = -__expf(A_log[d * NSTATE + n]);
  float Dv = Dp[d];

  float h[NSTATE];
#pragma unroll
  for (int n = 0; n < NSTATE; ++n) h[n] = 0.0f;

  for (int l = 0; l < LSEQ; ++l) {
    size_t m  = (size_t)b * LSEQ + l;
    float dt  = delta[m * DIN + d];
    float uu  = u[m * DIN + d];
    const float* Bc = xdbl + m * 48 + 16;
    const float* Cc = xdbl + m * 48 + 32;
    float dtu = dt * uu;
    float yv = 0.0f;
#pragma unroll
    for (int n = 0; n < NSTATE; ++n) {
      float dA = __expf(dt * An[n]);
      h[n] = dA * h[n] + dtu * Bc[n];
      yv += h[n] * Cc[n];
    }
    y[m * DIN + d] = yv + uu * Dv;
  }
}

// ---------------- gate: yg = y * silu(res), emit bf16 ----------------
__global__ void k_gate(const float* __restrict__ y,     // M x DIN
                       const float* __restrict__ xr,    // M x 2*DIN (res at +DIN)
                       __bf16* __restrict__ yg_bf) {
  int idx = blockIdx.x * blockDim.x + threadIdx.x;
  if (idx >= MROWS * DIN) return;
  int d = idx & (DIN - 1);
  int m = idx >> 9;
  float res = xr[(size_t)m * (2 * DIN) + DIN + d];
  yg_bf[idx] = f2bf(y[idx] * silu_f(res));
}

// ---------------- LayerNorm: one wave per row of 256 ----------------
__global__ void k_layernorm(const float* __restrict__ x,
                            const float* __restrict__ g,
                            const float* __restrict__ b,
                            float* __restrict__ out, int rows) {
  int row  = blockIdx.x * (blockDim.x >> 5) + (threadIdx.x >> 5);
  int lane = threadIdx.x & 31;
  if (row >= rows) return;
  const float* xr = x + (size_t)row * DMODEL;
  float s = 0.0f, s2 = 0.0f;
#pragma unroll
  for (int i = lane; i < DMODEL; i += 32) {
    float v = xr[i];
    s += v; s2 += v * v;
  }
#pragma unroll
  for (int off = 16; off > 0; off >>= 1) {
    s  += __shfl_down(s,  off, 32);
    s2 += __shfl_down(s2, off, 32);
  }
  s  = __shfl(s,  0, 32);
  s2 = __shfl(s2, 0, 32);
  float mu  = s * (1.0f / DMODEL);
  float var = s2 * (1.0f / DMODEL) - mu * mu;
  float inv = rsqrtf(var + 1e-5f);
#pragma unroll
  for (int i = lane; i < DMODEL; i += 32)
    out[(size_t)row * DMODEL + i] = (xr[i] - mu) * inv * g[i] + b[i];
}

// ---------------------------------------------------------------------------

static inline int gemm_blocks(int M, int N, int tnt) {
  int tiles = (M >> 5) * (N / (16 * tnt));
  return (tiles + 7) / 8;   // 8 waves (256 threads) per block
}

extern "C" void kernel_launch(void* const* d_in, const int* in_sizes, int n_in,
                              void* d_out, int out_size, void* d_ws, size_t ws_size,
                              hipStream_t stream) {
  (void)in_sizes; (void)n_in; (void)out_size; (void)ws_size;

  const float* x = (const float*)d_in[0];
  const float* P1[9]; const float* P2[9];
  for (int i = 0; i < 9; ++i) { P1[i] = (const float*)d_in[1 + i]; P2[i] = (const float*)d_in[10 + i]; }
  const float* g1    = (const float*)d_in[19];
  const float* beta1 = (const float*)d_in[20];
  const float* g2    = (const float*)d_in[21];
  const float* beta2 = (const float*)d_in[22];
  float* out = (float*)d_out;

  // ---- workspace carve-up (256B aligned) ----
  char* base = (char*)d_ws;
  size_t off = 0;
  auto alloc = [&](size_t bytes) -> void* {
    void* p = base + off;
    off += (bytes + 255) & ~(size_t)255;
    return p;
  };
  __bf16* act_bf  = (__bf16*)alloc((size_t)MROWS * DMODEL * 2);
  __bf16* win_bf  = (__bf16*)alloc((size_t)(2 * DIN) * DMODEL * 2);
  __bf16* wxp_bf  = (__bf16*)alloc((size_t)(DTRANK + 2 * NSTATE) * DIN * 2);
  __bf16* wout_bf = (__bf16*)alloc((size_t)DMODEL * DIN * 2);
  float*  xr      = (float*) alloc((size_t)MROWS * 2 * DIN * 4);
  float*  u       = (float*) alloc((size_t)MROWS * DIN * 4);
  __bf16* u_bf    = (__bf16*)alloc((size_t)MROWS * DIN * 2);
  float*  xdbl    = (float*) alloc((size_t)MROWS * 48 * 4);
  float*  deltaB  = (float*) alloc((size_t)MROWS * DIN * 4);
  float*  yB      = (float*) alloc((size_t)MROWS * DIN * 4);
  __bf16* yg_bf   = (__bf16*)alloc((size_t)MROWS * DIN * 2);
  float*  mo      = (float*) alloc((size_t)MROWS * DMODEL * 4);
  float*  h1      = (float*) alloc((size_t)MROWS * DMODEL * 4);

  const int T = 256;
  auto nb = [](int n) { return (n + 255) / 256; };

  // ===================== run one mamba block =====================
  auto run_block = [&](const float* act, const float* const* P,
                       const float* g, const float* beta, float* lnout) {
    const float* w_in  = P[0];
    const float* wconv = P[1];
    const float* bconv = P[2];
    const float* w_xp  = P[3];
    const float* w_dt  = P[4];
    const float* b_dt  = P[5];
    const float* A_log = P[6];
    const float* Dp    = P[7];
    const float* w_out = P[8];

    // converts
    k_cvt_bf16<<<nb(MROWS * DMODEL), T, 0, stream>>>(act, act_bf, MROWS * DMODEL);
    k_cvt_bf16<<<nb(2 * DIN * DMODEL), T, 0, stream>>>(w_in, win_bf, 2 * DIN * DMODEL);
    k_cvt_bf16<<<nb(48 * DIN), T, 0, stream>>>(w_xp, wxp_bf, 48 * DIN);
    k_cvt_bf16<<<nb(DMODEL * DIN), T, 0, stream>>>(w_out, wout_bf, DMODEL * DIN);

    // xr = act @ w_in^T   (M x 1024, K = 256), 32x32 wave tiles
    k_gemm_bf16_wmma<2><<<gemm_blocks(MROWS, 2 * DIN, 2), T, 0, stream>>>(
        act_bf, win_bf, xr, MROWS, 2 * DIN, DMODEL);

    // causal conv + silu -> u
    k_conv_silu<<<nb(MROWS * DIN), T, 0, stream>>>(xr, wconv, bconv, u, u_bf);

    // x_dbl = u @ w_xp^T   (M x 48, K = 512), 32x48 wave tiles
    k_gemm_bf16_wmma<3><<<gemm_blocks(MROWS, 48, 3), T, 0, stream>>>(
        u_bf, wxp_bf, xdbl, MROWS, 48, DIN);

    // delta = softplus(x_dbl[:, :16] @ w_dt^T + b_dt)
    k_delta<<<nb(MROWS * DIN), T, 0, stream>>>(xdbl, w_dt, b_dt, deltaB);

    // selective scan
    k_scan<<<nb(BATCH * DIN), T, 0, stream>>>(deltaB, u, xdbl, A_log, Dp, yB);

    // gate
    k_gate<<<nb(MROWS * DIN), T, 0, stream>>>(yB, xr, yg_bf);

    // out projection: mo = yg @ w_out^T   (M x 256, K = 512), 32x32 wave tiles
    k_gemm_bf16_wmma<2><<<gemm_blocks(MROWS, DMODEL, 2), T, 0, stream>>>(
        yg_bf, wout_bf, mo, MROWS, DMODEL, DIN);

    // layernorm (one wave per row)
    k_layernorm<<<(MROWS + 7) / 8, T, 0, stream>>>(mo, g, beta, lnout, MROWS);
  };

  run_block(x,  P1, g1, beta1, h1);    // block 1 -> h1
  run_block(h1, P2, g2, beta2, out);   // block 2 -> d_out
}
